// MLPGraphNetwork_38912403702319
// MI455X (gfx1250) — matrix operand order repaired and checked by reading, not compiled, not run on hardware
//
#include <hip/hip_runtime.h>
#include <hip/hip_bf16.h>

typedef _Float16 v16h __attribute__((ext_vector_type(16)));
typedef _Float16 v8h  __attribute__((ext_vector_type(8)));
typedef _Float16 v4h  __attribute__((ext_vector_type(4)));
typedef float    v8f  __attribute__((ext_vector_type(8)));
typedef float    v4f  __attribute__((ext_vector_type(4)));

#define WAVES 4          // 128 threads/block = 4 wave32
#define STG_STRIDE 200   // halves; 400B rows -> 16 distinct bank groups for b128 reads
#define HID_STRIDE 72    // halves; 144B rows -> conflict-free

__device__ __forceinline__ void atomAddF(float* p, float v) {
    __hip_atomic_fetch_add(p, v, __ATOMIC_RELAXED, __HIP_MEMORY_SCOPE_AGENT);
}

__device__ __forceinline__ void wait_ds() {
    asm volatile("s_wait_dscnt 0" ::: "memory");
}
__device__ __forceinline__ void wait_ld_ds() {
    asm volatile("s_wait_loadcnt 0\n\ts_wait_dscnt 0" ::: "memory");
}

// Build v16h A-fragment (16x32 f16 A layout, ISA 7.12.2) from LDS row.
__device__ __forceinline__ v16h load_afrag(const _Float16* base) {
    v8h lo = *(const v8h*)base;        // K group 0 (8 halves, 16B aligned)
    v8h hi = *(const v8h*)(base + 16); // K group 1
    v16h a;
#pragma unroll
    for (int i = 0; i < 8; ++i) { a[i] = lo[i]; a[i + 8] = hi[i]; }
    return a;
}

// ---------------------------------------------------------------------------
// prep: effective biases (fold g@W1[192:256]) + pack weights into B-fragment
// layout: flat[((kt*4+nt)*32+lane)*16 + e] = W[kt*32 + (lane<16?0:8) + (e&7) + (e&8)*2][nt*16 + lane%16]
// ---------------------------------------------------------------------------
__global__ void prep_kernel(const float* __restrict__ g,
                            const float* __restrict__ eW1, const float* __restrict__ eb1,
                            const float* __restrict__ eW2,
                            const float* __restrict__ nW1, const float* __restrict__ nb1,
                            const float* __restrict__ nW2,
                            float* __restrict__ ebias, float* __restrict__ nbias,
                            _Float16* __restrict__ w1pe, _Float16* __restrict__ w2pe,
                            _Float16* __restrict__ w1pn, _Float16* __restrict__ w2pn) {
    int t = threadIdx.x; // 256 threads
    if (t < 64) {
        float s = eb1[t];
        for (int k = 0; k < 64; ++k) s += g[k] * eW1[(192 + k) * 64 + t];
        ebias[t] = s;
    } else if (t < 128) {
        int n = t - 64;
        float s = nb1[n];
        for (int k = 0; k < 64; ++k) s += g[k] * nW1[(192 + k) * 64 + n];
        nbias[n] = s;
    }
    // W1: 6 k-tiles x 4 n-tiles
    for (int idx = t; idx < 6 * 4 * 32 * 16; idx += 256) {
        int e    = idx & 15;
        int lane = (idx >> 4) & 31;
        int nt   = (idx >> 9) & 3;
        int kt   = idx >> 11;
        int K = kt * 32 + ((lane < 16) ? 0 : 8) + (e & 7) + ((e & 8) << 1);
        int N = nt * 16 + (lane & 15);
        w1pe[idx] = (_Float16)eW1[K * 64 + N];
        w1pn[idx] = (_Float16)nW1[K * 64 + N];
    }
    // W2: 2 k-tiles x 4 n-tiles
    for (int idx = t; idx < 2 * 4 * 32 * 16; idx += 256) {
        int e    = idx & 15;
        int lane = (idx >> 4) & 31;
        int nt   = (idx >> 9) & 3;
        int kt   = idx >> 11;
        int K = kt * 32 + ((lane < 16) ? 0 : 8) + (e & 7) + ((e & 8) << 1);
        int N = nt * 16 + (lane & 15);
        w2pe[idx] = (_Float16)eW2[K * 64 + N];
        w2pn[idx] = (_Float16)nW2[K * 64 + N];
    }
}

// ---------------------------------------------------------------------------
// degree counts (float for cheap segment_mean divide)
// ---------------------------------------------------------------------------
__global__ void count_kernel(const int* __restrict__ senders, const int* __restrict__ receivers,
                             float* __restrict__ cnt_s, float* __restrict__ cnt_r, int n_edges) {
    int i = blockIdx.x * 256 + threadIdx.x;
    if (i < n_edges) {
        atomAddF(cnt_s + senders[i], 1.0f);
        atomAddF(cnt_r + receivers[i], 1.0f);
    }
}

// ---------------------------------------------------------------------------
// edge MLP: per wave, one tile of 16 edges. K=192 layer1, K=64 layer2.
// ---------------------------------------------------------------------------
__global__ void __launch_bounds__(WAVES * 32)
edge_kernel(const float* __restrict__ edges, const float* __restrict__ nodes,
            const int* __restrict__ senders, const int* __restrict__ receivers,
            const _Float16* __restrict__ w1p, const _Float16* __restrict__ w2p,
            const float* __restrict__ bias1, const float* __restrict__ bias2,
            float* __restrict__ out_edges, float* __restrict__ sent_sum,
            float* __restrict__ recv_sum, float* __restrict__ edge_colsum,
            int n_edges) {
    __shared__ _Float16 stag[WAVES][16 * STG_STRIDE];
    __shared__ _Float16 hid[WAVES][16 * HID_STRIDE];
    __shared__ int sidx[WAVES][16];
    __shared__ int ridx[WAVES][16];

    const int lane = threadIdx.x & 31;
    const int wv   = threadIdx.x >> 5;
    const int tile = blockIdx.x * WAVES + wv;
    const int e0   = tile * 16;
    if (e0 >= n_edges) return;

    if (lane < 16) {
        int e = e0 + lane;
        if (e >= n_edges) e = n_edges - 1;
        sidx[wv][lane] = senders[e];
        ridx[wv][lane] = receivers[e];
    }
    wait_ld_ds();

    _Float16* st = stag[wv];
    // gather [16 rows x 192 cols] f32 -> f16 LDS; 768 float4 chunks, 24/lane
#pragma unroll 4
    for (int c = lane; c < 768; c += 32) {
        int row = c / 48;
        int col = (c % 48) * 4;
        int e = e0 + row; if (e >= n_edges) e = n_edges - 1;
        const float* src;
        if (col < 64)        src = edges + (size_t)e * 64 + col;
        else if (col < 128)  src = nodes + (size_t)sidx[wv][row] * 64 + (col - 64);
        else                 src = nodes + (size_t)ridx[wv][row] * 64 + (col - 128);
        v4f v = *(const v4f*)src;
        v4h h = { (_Float16)v[0], (_Float16)v[1], (_Float16)v[2], (_Float16)v[3] };
        *(v4h*)(st + row * STG_STRIDE + col) = h;
    }
    wait_ld_ds();

    const int row  = lane & 15;
    const int koff = (lane < 16) ? 0 : 8;

    v16h afrag[6];
#pragma unroll
    for (int kt = 0; kt < 6; ++kt)
        afrag[kt] = load_afrag(st + row * STG_STRIDE + kt * 32 + koff);

    // ---- layer 1: 16x192 @ 192x64, ReLU, stash hidden in LDS (A layout src) ----
    const int Ncol  = lane & 15;
    const int mbase = (lane < 16) ? 0 : 8;
    _Float16* h = hid[wv];
#pragma unroll
    for (int nt = 0; nt < 4; ++nt) {
        float b = bias1[nt * 16 + Ncol];
        v8f acc = { b, b, b, b, b, b, b, b };
#pragma unroll
        for (int kt = 0; kt < 6; ++kt) {
            v16h w = *(const v16h*)(w1p + (size_t)(((kt << 2) | nt) * 32 + lane) * 16);
            acc = __builtin_amdgcn_wmma_f32_16x16x32_f16(false, afrag[kt], false, w,
                                                         (short)0, acc, false, false);
        }
        int N = nt * 16 + Ncol;
#pragma unroll
        for (int r = 0; r < 8; ++r) {
            float v = acc[r] > 0.0f ? acc[r] : 0.0f;
            h[(mbase + r) * HID_STRIDE + N] = (_Float16)v;
        }
    }
    wait_ds();

    v16h hfrag[2];
#pragma unroll
    for (int kt = 0; kt < 2; ++kt)
        hfrag[kt] = load_afrag(h + row * HID_STRIDE + kt * 32 + koff);

    // ---- layer 2: 16x64 @ 64x64, store + segment-sum scatter ----
#pragma unroll
    for (int nt = 0; nt < 4; ++nt) {
        float b = bias2[nt * 16 + Ncol];
        v8f acc = { b, b, b, b, b, b, b, b };
#pragma unroll
        for (int kt = 0; kt < 2; ++kt) {
            v16h w = *(const v16h*)(w2p + (size_t)(((kt << 2) | nt) * 32 + lane) * 16);
            acc = __builtin_amdgcn_wmma_f32_16x16x32_f16(false, hfrag[kt], false, w,
                                                         (short)0, acc, false, false);
        }
        int N = nt * 16 + Ncol;
        float csum = 0.0f;
#pragma unroll
        for (int r = 0; r < 8; ++r) {
            int M = mbase + r;
            int e = e0 + M;
            if (e < n_edges) {
                float v = acc[r];
                out_edges[(size_t)e * 64 + N] = v;
                atomAddF(sent_sum + (size_t)sidx[wv][M] * 64 + N, v);
                atomAddF(recv_sum + (size_t)ridx[wv][M] * 64 + N, v);
                csum += v;
            }
        }
        atomAddF(edge_colsum + N, csum);
    }
}

// ---------------------------------------------------------------------------
// node MLP: input = [node | sent_mean | recv_mean], K=192
// ---------------------------------------------------------------------------
__global__ void __launch_bounds__(WAVES * 32)
node_kernel(const float* __restrict__ nodes,
            const float* __restrict__ sent_sum, const float* __restrict__ recv_sum,
            const float* __restrict__ cnt_s, const float* __restrict__ cnt_r,
            const _Float16* __restrict__ w1p, const _Float16* __restrict__ w2p,
            const float* __restrict__ bias1, const float* __restrict__ bias2,
            float* __restrict__ out_nodes, float* __restrict__ node_colsum,
            int n_nodes) {
    __shared__ _Float16 stag[WAVES][16 * STG_STRIDE];
    __shared__ _Float16 hid[WAVES][16 * HID_STRIDE];
    __shared__ float rs[WAVES][16];
    __shared__ float rr[WAVES][16];

    const int lane = threadIdx.x & 31;
    const int wv   = threadIdx.x >> 5;
    const int tile = blockIdx.x * WAVES + wv;
    const int n0   = tile * 16;
    if (n0 >= n_nodes) return;

    if (lane < 16) {
        int n = n0 + lane; if (n >= n_nodes) n = n_nodes - 1;
        rs[wv][lane] = 1.0f / fmaxf(cnt_s[n], 1.0f);
        rr[wv][lane] = 1.0f / fmaxf(cnt_r[n], 1.0f);
    }
    wait_ld_ds();

    _Float16* st = stag[wv];
#pragma unroll 4
    for (int c = lane; c < 768; c += 32) {
        int row = c / 48;
        int col = (c % 48) * 4;
        int n = n0 + row; if (n >= n_nodes) n = n_nodes - 1;
        v4f v;
        if (col < 64) {
            v = *(const v4f*)(nodes + (size_t)n * 64 + col);
        } else if (col < 128) {
            v = *(const v4f*)(sent_sum + (size_t)n * 64 + (col - 64));
            float s = rs[wv][row];
            v[0] *= s; v[1] *= s; v[2] *= s; v[3] *= s;
        } else {
            v = *(const v4f*)(recv_sum + (size_t)n * 64 + (col - 128));
            float s = rr[wv][row];
            v[0] *= s; v[1] *= s; v[2] *= s; v[3] *= s;
        }
        v4h hh = { (_Float16)v[0], (_Float16)v[1], (_Float16)v[2], (_Float16)v[3] };
        *(v4h*)(st + row * STG_STRIDE + col) = hh;
    }
    wait_ld_ds();

    const int row  = lane & 15;
    const int koff = (lane < 16) ? 0 : 8;

    v16h afrag[6];
#pragma unroll
    for (int kt = 0; kt < 6; ++kt)
        afrag[kt] = load_afrag(st + row * STG_STRIDE + kt * 32 + koff);

    const int Ncol  = lane & 15;
    const int mbase = (lane < 16) ? 0 : 8;
    _Float16* h = hid[wv];
#pragma unroll
    for (int nt = 0; nt < 4; ++nt) {
        float b = bias1[nt * 16 + Ncol];
        v8f acc = { b, b, b, b, b, b, b, b };
#pragma unroll
        for (int kt = 0; kt < 6; ++kt) {
            v16h w = *(const v16h*)(w1p + (size_t)(((kt << 2) | nt) * 32 + lane) * 16);
            acc = __builtin_amdgcn_wmma_f32_16x16x32_f16(false, afrag[kt], false, w,
                                                         (short)0, acc, false, false);
        }
        int N = nt * 16 + Ncol;
#pragma unroll
        for (int r = 0; r < 8; ++r) {
            float v = acc[r] > 0.0f ? acc[r] : 0.0f;
            h[(mbase + r) * HID_STRIDE + N] = (_Float16)v;
        }
    }
    wait_ds();

    v16h hfrag[2];
#pragma unroll
    for (int kt = 0; kt < 2; ++kt)
        hfrag[kt] = load_afrag(h + row * HID_STRIDE + kt * 32 + koff);

#pragma unroll
    for (int nt = 0; nt < 4; ++nt) {
        float b = bias2[nt * 16 + Ncol];
        v8f acc = { b, b, b, b, b, b, b, b };
#pragma unroll
        for (int kt = 0; kt < 2; ++kt) {
            v16h w = *(const v16h*)(w2p + (size_t)(((kt << 2) | nt) * 32 + lane) * 16);
            acc = __builtin_amdgcn_wmma_f32_16x16x32_f16(false, hfrag[kt], false, w,
                                                         (short)0, acc, false, false);
        }
        int N = nt * 16 + Ncol;
        float csum = 0.0f;
#pragma unroll
        for (int r = 0; r < 8; ++r) {
            int n = n0 + mbase + r;
            if (n < n_nodes) {
                float v = acc[r];
                out_nodes[(size_t)n * 64 + N] = v;
                csum += v;
            }
        }
        atomAddF(node_colsum + N, csum);
    }
}

// ---------------------------------------------------------------------------
// global MLP: [node_mean | edge_mean | g] (192) -> 64 -> 64
// ---------------------------------------------------------------------------
__global__ void global_kernel(const float* __restrict__ node_colsum,
                              const float* __restrict__ edge_colsum,
                              const float* __restrict__ g,
                              const float* __restrict__ gW1, const float* __restrict__ gb1,
                              const float* __restrict__ gW2, const float* __restrict__ gb2,
                              float* __restrict__ out, float inv_n, float inv_e) {
    __shared__ float in3[192];
    __shared__ float h[64];
    int t = threadIdx.x; // 64 threads
    in3[t]       = node_colsum[t] * inv_n;
    in3[t + 64]  = edge_colsum[t] * inv_e;
    in3[t + 128] = g[t];
    __syncthreads();
    float s = gb1[t];
    for (int k = 0; k < 192; ++k) s += in3[k] * gW1[k * 64 + t];
    h[t] = s > 0.0f ? s : 0.0f;
    __syncthreads();
    float o = gb2[t];
    for (int k = 0; k < 64; ++k) o += h[k] * gW2[k * 64 + t];
    out[t] = o;
}

// ---------------------------------------------------------------------------
extern "C" void kernel_launch(void* const* d_in, const int* in_sizes, int n_in,
                              void* d_out, int out_size, void* d_ws, size_t ws_size,
                              hipStream_t stream) {
    const float* nodes   = (const float*)d_in[0];
    const float* edges   = (const float*)d_in[1];
    const float* g       = (const float*)d_in[2];
    const int*   senders = (const int*)d_in[3];
    const int*   recvers = (const int*)d_in[4];
    const float* eW1 = (const float*)d_in[5];
    const float* eb1 = (const float*)d_in[6];
    const float* eW2 = (const float*)d_in[7];
    const float* eb2 = (const float*)d_in[8];
    const float* nW1 = (const float*)d_in[9];
    const float* nb1 = (const float*)d_in[10];
    const float* nW2 = (const float*)d_in[11];
    const float* nb2 = (const float*)d_in[12];
    const float* gW1 = (const float*)d_in[13];
    const float* gb1 = (const float*)d_in[14];
    const float* gW2 = (const float*)d_in[15];
    const float* gb2 = (const float*)d_in[16];

    const int n_nodes = in_sizes[0] / 64;
    const int n_edges = in_sizes[1] / 64;

    // output layout: new_nodes | new_edges | new_globals
    float* out_nodes = (float*)d_out;
    float* out_edges = out_nodes + (size_t)n_nodes * 64;
    float* out_glob  = out_edges + (size_t)n_edges * 64;

    // workspace layout (f32 accumulators first = one contiguous memset region)
    float* ws_f = (float*)d_ws;
    float* sent_sum    = ws_f;                                  // N*64
    float* recv_sum    = sent_sum + (size_t)n_nodes * 64;       // N*64
    float* cnt_s       = recv_sum + (size_t)n_nodes * 64;       // N
    float* cnt_r       = cnt_s + n_nodes;                       // N
    float* node_colsum = cnt_r + n_nodes;                       // 64
    float* edge_colsum = node_colsum + 64;                      // 64
    size_t zero_floats = (size_t)n_nodes * 64 * 2 + (size_t)n_nodes * 2 + 128;
    float* ebias = edge_colsum + 64;                            // 64
    float* nbias = ebias + 64;                                  // 64
    _Float16* w1pe = (_Float16*)(nbias + 64);                   // 12288 halves
    _Float16* w2pe = w1pe + 6 * 4 * 32 * 16;                    // 4096
    _Float16* w1pn = w2pe + 2 * 4 * 32 * 16;                    // 12288
    _Float16* w2pn = w1pn + 6 * 4 * 32 * 16;                    // 4096

    hipMemsetAsync(d_ws, 0, zero_floats * sizeof(float), stream);

    prep_kernel<<<1, 256, 0, stream>>>(g, eW1, eb1, eW2, nW1, nb1, nW2,
                                       ebias, nbias, w1pe, w2pe, w1pn, w2pn);

    count_kernel<<<(n_edges + 255) / 256, 256, 0, stream>>>(senders, recvers,
                                                            cnt_s, cnt_r, n_edges);

    int e_tiles = (n_edges + 15) / 16;
    edge_kernel<<<(e_tiles + WAVES - 1) / WAVES, WAVES * 32, 0, stream>>>(
        edges, nodes, senders, recvers, w1pe, w2pe, ebias, eb2,
        out_edges, sent_sum, recv_sum, edge_colsum, n_edges);

    int n_tiles = (n_nodes + 15) / 16;
    node_kernel<<<(n_tiles + WAVES - 1) / WAVES, WAVES * 32, 0, stream>>>(
        nodes, sent_sum, recv_sum, cnt_s, cnt_r, w1pn, w2pn, nbias, nb2,
        out_nodes, node_colsum, n_nodes);

    global_kernel<<<1, 64, 0, stream>>>(node_colsum, edge_colsum, g,
                                        gW1, gb1, gW2, gb2, out_glob,
                                        1.0f / (float)n_nodes, 1.0f / (float)n_edges);
}
